// RelativeMultiHeadAttn_34033320853583
// MI455X (gfx1250) — matrix operand order repaired and checked by reading, not compiled, not run on hardware
//
#include <hip/hip_runtime.h>

// ---------- types & helpers ----------
typedef __attribute__((ext_vector_type(16))) __bf16 bf16x16;
typedef __attribute__((ext_vector_type(8)))  __bf16 bf16x8;
typedef __attribute__((ext_vector_type(8)))  float  f32x8;
typedef __attribute__((ext_vector_type(4)))  float  f32x4;

__device__ __forceinline__ bf16x16 combine8(bf16x8 lo, bf16x8 hi) {
    bf16x16 r;
    #pragma unroll
    for (int i = 0; i < 8; ++i) { r[i] = lo[i]; r[i + 8] = hi[i]; }
    return r;
}

// A-fragment (16-bit, 16x32): element t of lane-half h covers K below
__device__ __forceinline__ int kidxA(int t, int h) {
    return (t < 8) ? (h * 8 + t) : (16 + h * 8 + (t - 8));
}
// B-fragment (16-bit, 32x16): same K order as A-fragment
__device__ __forceinline__ int kidxB(int t, int h) { return (t < 8) ? (h * 8 + t) : (16 + h * 8 + (t - 8)); }

__device__ __forceinline__ f32x8 wmma_bf16(bf16x16 a, bf16x16 b, f32x8 c) {
    return __builtin_amdgcn_wmma_f32_16x16x32_bf16(false, a, false, b, (short)0, c, false, false);
}

// ---------- generic f32 GEMM via bf16 WMMA: C[M,N] = A[M,K]*B[K,N] (+resid) ----------
__global__ __launch_bounds__(256) void gemm_f32_wmma(
    const float* __restrict__ A, const float* __restrict__ B,
    const float* __restrict__ resid, float* __restrict__ C,
    int M, int N, int K)
{
    const int lane = threadIdx.x & 31;
    const int wave = threadIdx.x >> 5;
    const int wm = wave >> 2, wn = wave & 3;            // 2x4 wave grid
    const int row0 = blockIdx.y * 32 + wm * 16;
    const int col0 = blockIdx.x * 64 + wn * 16;
    const int m = lane & 15, h = lane >> 4;

    f32x8 acc = {};
    const float* arow = A + (size_t)(row0 + m) * K;
    for (int kc = 0; kc < K; kc += 32) {
        __builtin_prefetch(arow + kc + 64, 0, 1);
        f32x4 a0 = *(const f32x4*)(arow + kc + h * 8);
        f32x4 a1 = *(const f32x4*)(arow + kc + h * 8 + 4);
        f32x4 a2 = *(const f32x4*)(arow + kc + 16 + h * 8);
        f32x4 a3 = *(const f32x4*)(arow + kc + 16 + h * 8 + 4);
        bf16x16 av;
        #pragma unroll
        for (int t = 0; t < 4; ++t) {
            av[t]      = (__bf16)a0[t];
            av[4 + t]  = (__bf16)a1[t];
            av[8 + t]  = (__bf16)a2[t];
            av[12 + t] = (__bf16)a3[t];
        }
        bf16x16 bv;
        const float* bcol = B + (size_t)(kc + h * 8) * N + (col0 + m);
        #pragma unroll
        for (int t = 0; t < 16; ++t) bv[t] = (__bf16)bcol[(size_t)(t + (t & 8)) * N];
        acc = wmma_bf16(av, bv, acc);
    }
    const int cc = col0 + m;
    const size_t cbase = (size_t)(row0 + h * 8) * N + cc;
    if (resid != nullptr) {
        #pragma unroll
        for (int r = 0; r < 8; ++r)
            C[cbase + (size_t)r * N] = acc[r] + resid[cbase + (size_t)r * N];
    } else {
        #pragma unroll
        for (int r = 0; r < 8; ++r)
            C[cbase + (size_t)r * N] = acc[r];
    }
}

// ---------- pack kernels: build bf16 operands in fragment-native layouts ----------
// qrw/qrr: [bn][2048][64] row-major (A-side)
__global__ void pack_q_kernel(const float* __restrict__ heads,
                              const float* __restrict__ rwb, const float* __restrict__ rrb,
                              __bf16* __restrict__ qrw, __bf16* __restrict__ qrr)
{
    int idx = blockIdx.x * blockDim.x + threadIdx.x;    // 2048*2*16*64
    if (idx >= (1 << 22)) return;
    int d = idx & 63, n = (idx >> 6) & 15, b = (idx >> 10) & 1, i = idx >> 11;
    float q = heads[((size_t)(i * 2 + b)) * 3072 + n * 64 + d];
    size_t o = ((size_t)(b * 16 + n) * 2048 + i) * 64 + d;
    qrw[o] = (__bf16)(q + rwb[n * 64 + d]);
    qrr[o] = (__bf16)(q + rrb[n * 64 + d]);
}

// kPack: [bn][jt=128][kc=2][lane=32][t=16]  (B fragments of k^T)
__global__ void pack_k_kernel(const float* __restrict__ heads, __bf16* __restrict__ kP)
{
    int idx = blockIdx.x * blockDim.x + threadIdx.x;
    if (idx >= (1 << 22)) return;
    int t = idx & 15, lane = (idx >> 4) & 31, kc = (idx >> 9) & 1;
    int jt = (idx >> 10) & 127, bn = idx >> 17;
    int j = jt * 16 + (lane & 15);
    int d = kc * 32 + kidxB(t, lane >> 4);
    int b = bn >> 4, n = bn & 15;
    kP[idx] = (__bf16)heads[((size_t)(j * 2 + b)) * 3072 + 1024 + n * 64 + d];
}

// vPack: [bn][dt=4][jc=64][lane=32][t=16]  (B fragments of v)
__global__ void pack_v_kernel(const float* __restrict__ heads, __bf16* __restrict__ vP)
{
    int idx = blockIdx.x * blockDim.x + threadIdx.x;
    if (idx >= (1 << 22)) return;
    int t = idx & 15, lane = (idx >> 4) & 31, jc = (idx >> 9) & 63;
    int dt = (idx >> 15) & 3, bn = idx >> 17;
    int d = dt * 16 + (lane & 15);
    int j = jc * 32 + kidxB(t, lane >> 4);
    int b = bn >> 4, n = bn & 15;
    vP[idx] = (__bf16)heads[((size_t)(j * 2 + b)) * 3072 + 2048 + n * 64 + d];
}

// rkPack: [n=16][jt=128][kc=2][lane=32][t=16]
__global__ void pack_rk_kernel(const float* __restrict__ rk, __bf16* __restrict__ rkP)
{
    int idx = blockIdx.x * blockDim.x + threadIdx.x;
    if (idx >= (1 << 21)) return;
    int t = idx & 15, lane = (idx >> 4) & 31, kc = (idx >> 9) & 1;
    int jt = (idx >> 10) & 127, n = idx >> 17;
    int j = jt * 16 + (lane & 15);
    int d = kc * 32 + kidxB(t, lane >> 4);
    rkP[idx] = (__bf16)rk[(size_t)j * 1024 + n * 64 + d];
}

// rel-shift pad zeros: BD[i, i+1, b, n] = 0 for i < 2047
__global__ void bd_zero_kernel(float* __restrict__ BD)
{
    int idx = blockIdx.x * blockDim.x + threadIdx.x;
    if (idx >= 2047 * 32) return;
    int bn = idx & 31, i = idx >> 5;
    BD[((size_t)i * 2048 + i + 1) * 32 + bn] = 0.0f;
}

// ---------- AC & shifted BD: one wave = one 16x16 tile, K=64 (2 WMMAs each) ----
// B operands staged global->LDS via async DMA (ASYNCcnt), then ds_load_b128 reads.
__global__ __launch_bounds__(256) void score_kernel(
    const __bf16* __restrict__ qrw, const __bf16* __restrict__ qrr,
    const __bf16* __restrict__ kP,  const __bf16* __restrict__ rkP,
    float* __restrict__ AC, float* __restrict__ BD)
{
    __shared__ __bf16 sK[8][2][32][16];   // 16 KB
    __shared__ __bf16 sR[8][2][32][16];   // 16 KB
    const int lane = threadIdx.x & 31;
    const int wave = threadIdx.x >> 5;
    const int jt = blockIdx.x * 8 + wave;   // 0..127
    const int it = blockIdx.y;              // 0..127
    const int bn = blockIdx.z;              // 0..31
    const int n  = bn & 15;
    const int m = lane & 15, h = lane >> 4;

    // per-lane async DMA: each lane pulls its own 32B fragment chunk (x2 kc) into LDS
    #pragma unroll
    for (int kc = 0; kc < 2; ++kc) {
        const __bf16* kb = kP  + (((size_t)bn * 128 + jt) * 2 + kc) * 512 + lane * 16;
        const __bf16* rb = rkP + (((size_t)n  * 128 + jt) * 2 + kc) * 512 + lane * 16;
        unsigned ldk = (unsigned)(unsigned long long)(&sK[wave][kc][lane][0]);
        unsigned ldr = (unsigned)(unsigned long long)(&sR[wave][kc][lane][0]);
        unsigned long long gk = (unsigned long long)kb;
        unsigned long long gr = (unsigned long long)rb;
        asm volatile(
            "global_load_async_to_lds_b128 %0, %1, off\n\t"
            "global_load_async_to_lds_b128 %0, %1, off offset:16\n\t"
            "global_load_async_to_lds_b128 %2, %3, off\n\t"
            "global_load_async_to_lds_b128 %2, %3, off offset:16"
            :: "v"(ldk), "v"(gk), "v"(ldr), "v"(gr) : "memory");
    }

    const __bf16* qw = qrw + ((size_t)bn * 2048 + it * 16 + m) * 64;
    const __bf16* qr = qrr + ((size_t)bn * 2048 + it * 16 + m) * 64;

    asm volatile("s_wait_asynccnt 0x0" ::: "memory");

    f32x8 accAC = {}, accBD = {};
    #pragma unroll
    for (int kc = 0; kc < 2; ++kc) {
        bf16x16 aw = combine8(*(const bf16x8*)(qw + kc * 32 + h * 8),
                              *(const bf16x8*)(qw + kc * 32 + 16 + h * 8));
        bf16x16 ar = combine8(*(const bf16x8*)(qr + kc * 32 + h * 8),
                              *(const bf16x8*)(qr + kc * 32 + 16 + h * 8));
        bf16x16 bk = combine8(*(const bf16x8*)&sK[wave][kc][lane][0],
                              *(const bf16x8*)&sK[wave][kc][lane][8]);
        bf16x16 br = combine8(*(const bf16x8*)&sR[wave][kc][lane][0],
                              *(const bf16x8*)&sR[wave][kc][lane][8]);
        accAC = wmma_bf16(aw, bk, accAC);
        accBD = wmma_bf16(ar, br, accBD);
    }
    const int j = jt * 16 + m;
    #pragma unroll
    for (int r = 0; r < 8; ++r) {
        int i = it * 16 + h * 8 + r;
        AC[((size_t)i * 2048 + j) * 32 + bn] = accAC[r];
        int g = i * 2049 + j - 2047;                 // rel_shift scatter map
        if (g >= 0) BD[(size_t)g * 32 + bn] = accBD[r];
    }
}

// ---------- softmax over k axis; lane<->(b,n) keeps every access a 128B burst ----------
__global__ __launch_bounds__(256) void softmax_kernel(
    const float* __restrict__ AC, const float* __restrict__ BD, float* __restrict__ P)
{
    const int i  = blockIdx.x;
    const int bn = threadIdx.x & 31;
    const int jt = threadIdx.x >> 5;                 // 0..7
    __shared__ float sm[8][32];
    __shared__ float ss[8][32];

    const size_t rb = (size_t)i * 2048 * 32 + bn;
    float m = -3.4e38f, s = 0.f;
    for (int j = jt; j < 2048; j += 8) {
        float x = (AC[rb + (size_t)j * 32] + BD[rb + (size_t)j * 32]) * 0.125f;
        float mn = fmaxf(m, x);
        s = s * __expf(m - mn) + __expf(x - mn);
        m = mn;
    }
    sm[jt][bn] = m; ss[jt][bn] = s;
    __syncthreads();
    if (jt == 0) {
        float M = sm[0][bn], S = ss[0][bn];
        #pragma unroll
        for (int t = 1; t < 8; ++t) {
            float m2 = sm[t][bn], s2 = ss[t][bn];
            float mn = fmaxf(M, m2);
            S = S * __expf(M - mn) + s2 * __expf(m2 - mn);
            M = mn;
        }
        sm[0][bn] = M; ss[0][bn] = 1.f / S;
    }
    __syncthreads();
    float M = sm[0][bn], invS = ss[0][bn];
    for (int j = jt; j < 2048; j += 8) {
        float x = (AC[rb + (size_t)j * 32] + BD[rb + (size_t)j * 32]) * 0.125f;
        P[rb + (size_t)j * 32] = __expf(x - M) * invS;
    }
}

// ---------- attn_vec = prob @ v  (per bn: M=2048,N=64,K=2048) ----------
__global__ __launch_bounds__(256) void pv_kernel(
    const float* __restrict__ P, const __bf16* __restrict__ vP,
    float* __restrict__ avec)
{
    const int lane = threadIdx.x & 31;
    const int wave = threadIdx.x >> 5;
    const int it = blockIdx.y * 8 + wave;  // 0..127
    const int dt = blockIdx.x;             // 0..3
    const int bn = blockIdx.z;             // 0..31
    const int b = bn >> 4, n = bn & 15;
    const int m = lane & 15, h = lane >> 4;

    f32x8 acc = {};
    const int i = it * 16 + m;
    for (int jc = 0; jc < 64; ++jc) {
        bf16x16 av;
        #pragma unroll
        for (int t = 0; t < 16; ++t) {
            int j = jc * 32 + kidxA(t, h);
            av[t] = (__bf16)P[((size_t)i * 2048 + j) * 32 + bn];
        }
        const __bf16* vb = vP + (((size_t)bn * 4 + dt) * 64 + jc) * 512 + lane * 16;
        bf16x16 bv = combine8(*(const bf16x8*)vb, *(const bf16x8*)(vb + 8));
        acc = wmma_bf16(av, bv, acc);
    }
    const int d = dt * 16 + m;
    #pragma unroll
    for (int r = 0; r < 8; ++r) {
        int ig = it * 16 + h * 8 + r;
        avec[((size_t)(ig * 2 + b)) * 1024 + n * 64 + d] = acc[r];
    }
}

// ---------- host launcher ----------
extern "C" void kernel_launch(void* const* d_in, const int* in_sizes, int n_in,
                              void* d_out, int out_size, void* d_ws, size_t ws_size,
                              hipStream_t stream)
{
    const float* w      = (const float*)d_in[0];   // [2048,2,1024]
    const float* r      = (const float*)d_in[1];   // [2048,1024]
    const float* qkv_w  = (const float*)d_in[2];   // [1024,3072]
    const float* rnet_w = (const float*)d_in[3];   // [1024,1024]
    const float* o_w    = (const float*)d_in[4];   // [1024,1024]
    const float* rrb    = (const float*)d_in[5];   // r_r_bias [16,64]
    const float* rwb    = (const float*)d_in[6];   // r_w_bias [16,64]

    char* ws = (char*)d_ws;
    float* heads  = (float*)(ws);                    // 50,331,648 B
    float* rk     = (float*)(ws + 50331648);         //  8,388,608 B
    float* avec   = (float*)(ws + 58720256);         // 16,777,216 B
    __bf16* qrw   = (__bf16*)(ws + 75497472);        //  8,388,608 B
    __bf16* qrr   = (__bf16*)(ws + 83886080);        //  8,388,608 B
    __bf16* kP    = (__bf16*)(ws + 92274688);        //  8,388,608 B
    __bf16* vP    = (__bf16*)(ws + 100663296);       //  8,388,608 B
    __bf16* rkP   = (__bf16*)(ws + 109051904);       //  4,194,304 B

    float* out0 = (float*)d_out;                  // [2048,2,1024]
    float* PROB = out0 + 4194304;                 // [2048,2048,2,16]
    float* AC   = PROB + 134217728;
    float* BD   = AC   + 134217728;

    // 1) projections
    gemm_f32_wmma<<<dim3(3072 / 64, 4096 / 32), 256, 0, stream>>>(w, qkv_w, nullptr, heads, 4096, 3072, 1024);
    gemm_f32_wmma<<<dim3(1024 / 64, 2048 / 32), 256, 0, stream>>>(r, rnet_w, nullptr, rk, 2048, 1024, 1024);
    // 2) bf16 operand packing
    pack_q_kernel <<<(1 << 22) / 256, 256, 0, stream>>>(heads, rwb, rrb, qrw, qrr);
    pack_k_kernel <<<(1 << 22) / 256, 256, 0, stream>>>(heads, kP);
    pack_v_kernel <<<(1 << 22) / 256, 256, 0, stream>>>(heads, vP);
    pack_rk_kernel<<<(1 << 21) / 256, 256, 0, stream>>>(rk, rkP);
    // 3) scores (AC direct, BD through rel-shift scatter) + pad zeros
    bd_zero_kernel<<<(2047 * 32 + 255) / 256, 256, 0, stream>>>(BD);
    score_kernel<<<dim3(16, 128, 32), 256, 0, stream>>>(qrw, qrr, kP, rkP, AC, BD);
    // 4) softmax over k
    softmax_kernel<<<2048, 256, 0, stream>>>(AC, BD, PROB);
    // 5) prob @ v
    pv_kernel<<<dim3(4, 16, 32), 256, 0, stream>>>(PROB, vP, avec);
    // 6) out = w + attn_vec @ o_w
    gemm_f32_wmma<<<dim3(1024 / 64, 4096 / 32), 256, 0, stream>>>(avec, o_w, w, out0, 4096, 1024, 1024);
}
